// StyleGAN3Generator_13932873909229
// MI455X (gfx1250) — matrix operand order repaired and checked by reading, not compiled, not run on hardware
//
#include <hip/hip_runtime.h>
#include <cstdint>

// ---------------------------------------------------------------------------
// Types for CDNA5 WMMA
// ---------------------------------------------------------------------------
typedef __attribute__((ext_vector_type(2)))  float  v2f;
typedef __attribute__((ext_vector_type(8)))  float  v8f;
typedef __attribute__((ext_vector_type(16))) __bf16 v16bf;

#if defined(__HIP_DEVICE_COMPILE__) && defined(__has_builtin)
#if __has_builtin(__builtin_amdgcn_wmma_f32_16x16x4_f32)
#define USE_WMMA_F32 1
#endif
#endif

struct MapParams {
    const float* w[8];
    const float* b[8];
};

// ---------------------------------------------------------------------------
// Mapping network: z[8,512] -> w[8,512].
// ---------------------------------------------------------------------------
__global__ __launch_bounds__(1024) void mapping_kernel(
    const float* __restrict__ z, MapParams mp, float* __restrict__ wout)
{
    __shared__ float xb[16][512];
    __shared__ float scale[16];
    const int tid = threadIdx.x;

    for (int i = tid; i < 16 * 512; i += 1024) {
        int r = i >> 9, c = i & 511;
        xb[r][c] = (r < 8) ? z[r * 512 + c] : 0.f;
    }
    __syncthreads();
    if (tid < 16) {
        float s = 0.f;
        if (tid < 8) {
            for (int c = 0; c < 512; ++c) { float v = xb[tid][c]; s += v * v; }
            scale[tid] = rsqrtf(s * (1.f / 512.f) + 1e-8f);
        } else scale[tid] = 0.f;
    }
    __syncthreads();
    for (int i = tid; i < 16 * 512; i += 1024) {
        int r = i >> 9, c = i & 511;
        xb[r][c] *= scale[r];
    }
    __syncthreads();

    const float gain = 0.01f / 22.62741699796952f;   // LR_MULT / sqrt(512)
    const int lane = tid & 31;
    const int wv   = tid >> 5;
    const int n0   = wv * 16;
    const int m    = lane & 15;
    const int kb   = (lane >> 4) << 1;

    for (int layer = 0; layer < 8; ++layer) {
        const float* W  = mp.w[layer];
        const float* Bi = mp.b[layer];
#if defined(USE_WMMA_F32)
        v8f acc = {};
        for (int k0 = 0; k0 < 512; k0 += 4) {
            v2f a, b;
            a.x = xb[m][k0 + kb];
            a.y = xb[m][k0 + kb + 1];
            b.x = W[(n0 + m) * 512 + k0 + kb] * gain;
            b.y = W[(n0 + m) * 512 + k0 + kb + 1] * gain;
            acc = __builtin_amdgcn_wmma_f32_16x16x4_f32(
                false, a, false, b, (short)0, acc, false, false);
        }
        __syncthreads();
        const int rbase = (lane >> 4) << 3;
        #pragma unroll
        for (int r = 0; r < 8; ++r) {
            float v = acc[r] + Bi[n0 + m] * 0.01f;
            v = (v > 0.f) ? v : 0.2f * v;
            xb[rbase + r][n0 + m] = v;
        }
#else
        float tmp[4];
        #pragma unroll
        for (int j = 0; j < 4; ++j) {
            int i = tid + j * 1024;
            int rr = i >> 9, col = i & 511;
            float s = Bi[col] * 0.01f;
            for (int k = 0; k < 512; ++k)
                s += xb[rr][k] * W[col * 512 + k] * gain;
            tmp[j] = (s > 0.f) ? s : 0.2f * s;
        }
        __syncthreads();
        #pragma unroll
        for (int j = 0; j < 4; ++j) {
            int i = tid + j * 1024;
            xb[i >> 9][i & 511] = tmp[j];
        }
#endif
        __syncthreads();
    }
    for (int i = tid; i < 8 * 512; i += 1024)
        wout[i] = xb[i >> 9][i & 511];
}

// ---------------------------------------------------------------------------
// style[b, ci] = w[b] . aw[ci] + ab[ci]
// ---------------------------------------------------------------------------
__global__ __launch_bounds__(256) void style_kernel(
    const float* __restrict__ wvec, const float* __restrict__ aw,
    const float* __restrict__ ab, float* __restrict__ style, int cin)
{
    int i = blockIdx.x * 256 + threadIdx.x;
    if (i >= 8 * cin) return;
    int b = i / cin, ci = i % cin;
    float s = ab[ci];
    const float* wr = wvec + b * 512;
    const float* ar = aw + (long)ci * 512;
    for (int k = 0; k < 512; ++k) s += wr[k] * ar[k];
    style[i] = s;
}

// ---------------------------------------------------------------------------
// OIHW f32 -> tap-major [9][Cout][Cin] bf16
// ---------------------------------------------------------------------------
__global__ __launch_bounds__(256) void prep_weights_kernel(
    const float* __restrict__ w, __bf16* __restrict__ wt, int Cout, int Cin)
{
    int i = blockIdx.x * 256 + threadIdx.x;
    int total = Cout * Cin * 9;
    if (i >= total) return;
    int tap = i % 9;
    int t   = i / 9;
    int ci  = t % Cin;
    int co  = t / Cin;
    wt[(long)tap * Cout * Cin + (long)co * Cin + ci] = (__bf16)w[i];
}

// ---------------------------------------------------------------------------
// Zero the 1-pixel halo border of the NHWC bf16 activation buffer
// ---------------------------------------------------------------------------
__global__ __launch_bounds__(256) void fill_border_kernel(
    __bf16* __restrict__ xm, int Cin, int H, int W)
{
    int pc = 2 * (W + 2) + 2 * H;
    long total = 8L * Cin * pc;
    long i = (long)blockIdx.x * 256 + threadIdx.x;
    if (i >= total) return;
    int ci = (int)(i % Cin);
    long t = i / Cin;
    int cell = (int)(t % pc);
    int b = (int)(t / pc);
    int y, x;
    if (cell < W + 2)            { y = 0;     x = cell; }
    else if (cell < 2 * (W + 2)) { y = H + 1; x = cell - (W + 2); }
    else { int r2 = cell - 2 * (W + 2); y = 1 + (r2 >> 1); x = (r2 & 1) ? (W + 1) : 0; }
    xm[(((long)b * (H + 2) + y) * (W + 2) + x) * Cin + ci] = (__bf16)0.f;
}

// ---------------------------------------------------------------------------
// Pre-modulation (style * x -> bf16), NCHW f32 -> NHWC(+halo) bf16.
// ---------------------------------------------------------------------------
__global__ __launch_bounds__(256) void premod_plain_kernel(
    const float* __restrict__ xin, long xbs,
    const float* __restrict__ style, __bf16* __restrict__ xm,
    int Cin, int H, int W)
{
    __shared__ float rbuf[32][129];
    int row = blockIdx.x;
    int b = row / H, y = row % H;
    int ci0 = blockIdx.y << 5;
    int tid = threadIdx.x;
    int HW = H * W;
    for (int e = tid; e < 32 * W; e += 256) {
        int x = e % W, cl = e / W;
        rbuf[cl][x] = xin[b * xbs + (long)(ci0 + cl) * HW + y * W + x]
                    * style[b * Cin + ci0 + cl];
    }
    __syncthreads();
    long base = (((long)b * (H + 2) + (y + 1)) * (W + 2) + 1) * Cin + ci0;
    for (int e = tid; e < 32 * W; e += 256) {
        int cl = e & 31, x = e >> 5;
        xm[base + (long)x * Cin + cl] = (__bf16)rbuf[cl][x];
    }
}

// ---------------------------------------------------------------------------
// Fused bilinear x2 upsample + modulation + bf16, NCHW(H/2) -> NHWC(+halo) H.
// ---------------------------------------------------------------------------
__global__ __launch_bounds__(256) void premod_up_kernel(
    const float* __restrict__ xin,   // NCHW at H/2 x W/2
    const float* __restrict__ style, __bf16* __restrict__ xm,
    int Cin, int H, int W)
{
    __shared__ float r0[32][65], r1[32][65];
    int row = blockIdx.x;
    int b = row / H, y = row % H;
    int ci0 = blockIdx.y << 5;
    int tid = threadIdx.x;
    int H2 = H >> 1, W2 = W >> 1;
    float cy = y * 0.5f - 0.25f;
    int y0 = (int)floorf(cy);
    float fy = cy - y0;
    int y1 = min(max(y0 + 1, 0), H2 - 1);
    y0 = min(max(y0, 0), H2 - 1);
    long pb = (long)b * Cin * H2 * W2;
    for (int e = tid; e < 64 * W2; e += 256) {
        int x2 = e % W2;
        int t = e / W2;
        int wh = t & 1, cl = t >> 1;
        float v = xin[pb + (long)(ci0 + cl) * H2 * W2 + (wh ? y1 : y0) * W2 + x2]
                * style[b * Cin + ci0 + cl];
        if (wh) r1[cl][x2] = v; else r0[cl][x2] = v;
    }
    __syncthreads();
    long base = (((long)b * (H + 2) + (y + 1)) * (W + 2) + 1) * Cin + ci0;
    for (int e = tid; e < 32 * W; e += 256) {
        int cl = e & 31, x = e >> 5;
        float cx = x * 0.5f - 0.25f;
        int x0 = (int)floorf(cx);
        float fx = cx - x0;
        int x1 = min(max(x0 + 1, 0), W2 - 1);
        x0 = min(max(x0, 0), W2 - 1);
        float v = (1.f - fy) * ((1.f - fx) * r0[cl][x0] + fx * r0[cl][x1])
                +        fy  * ((1.f - fx) * r1[cl][x0] + fx * r1[cl][x1]);
        xm[base + (long)x * Cin + cl] = (__bf16)v;
    }
}

// ---------------------------------------------------------------------------
// 3x3 conv as implicit GEMM, bf16 WMMA 16x16x32, f32 accumulate.
// Block 256 thr = 8 waves, tile 64(M) x 64(N); each wave: 2 WMMAs / chunk.
// A/B staging uses GLOBAL_LOAD_ASYNC_TO_LDS_B128 (ASYNCcnt path, 16B/lane,
// no VGPR round-trip); triple-buffered LDS so ONE barrier per K-chunk is
// sufficient (a buffer is rewritten 3 chunks after its last read).
// s_wait_asynccnt 2 keeps exactly the next chunk's two loads in flight.
// ---------------------------------------------------------------------------
__global__ __launch_bounds__(256) void modconv3x3_kernel(
    const __bf16* __restrict__ xm,   // [B][H+2][W+2][Cin]
    const __bf16* __restrict__ wt,   // [9][Cout][Cin]
    const float* __restrict__ bias,
    float* __restrict__ out,         // NCHW f32
    int Cin, int Cout, int H, int W)
{
    __shared__ __attribute__((aligned(16))) __bf16 Ab[3][64][40];
    __shared__ __attribute__((aligned(16))) __bf16 Bb[3][64][40];
    __shared__ int nb[64], ny[64], nx[64];

    const int tid   = threadIdx.x;
    const int lane  = tid & 31;
    const int wv    = tid >> 5;
    const int mi    = wv >> 2;        // 0..1 : M rows [mi*32, mi*32+32)
    const int ni    = wv & 3;         // 0..3 : N cols [ni*16, ni*16+16)
    const int khalf = lane >> 4;
    const int col   = lane & 15;
    const int n0    = blockIdx.x * 64;
    const int co0   = blockIdx.y * 64;
    const int HW    = H * W;
    const int Ntot  = 8 * HW;
    const int Wp    = W + 2;
    const int nci   = Cin >> 5;
    const int nchunks = 9 * nci;
    const long CoutCin = (long)Cout * Cin;

    if (tid < 64) {
        int n = n0 + tid;
        int b = 0, y = 0, x = 0;
        if (n < Ntot) { b = n / HW; int px = n % HW; y = px / W; x = px % W; }
        nb[tid] = b; ny[tid] = y; nx[tid] = x;
    }
    __syncthreads();

    const int ro  = tid >> 2;
    const int seg = (tid & 3) << 3;     // bf16 element offset: 0,8,16,24
    const int rb  = nb[ro];
    const int ryy = ny[ro];
    const int rxx = nx[ro];
    // LDS byte addresses for this thread's 16B staging slot (buffer 0);
    // generic-pointer low 32 bits == LDS address (ISA: LDS aperture addr[31:0])
    const unsigned aBase = (unsigned)(uintptr_t)(void*)&Ab[0][ro][seg];
    const unsigned bBase = (unsigned)(uintptr_t)(void*)&Bb[0][ro][seg];
    const unsigned bufStride = 64u * 40u * 2u;   // 5120 B per buffer

    auto stageAsync = [&](int chunk, int s) {
        int tap = chunk / nci;
        int ci0 = (chunk - tap * nci) << 5;
        int ky = tap / 3;
        int kx = tap - ky * 3;          // halo-shifted: 0..2
        const __bf16* gA = wt + (long)tap * CoutCin + (long)(co0 + ro) * Cin + ci0 + seg;
        const __bf16* gB = xm + (((long)rb * (H + 2) + (ryy + ky)) * Wp + (rxx + kx)) * Cin
                              + ci0 + seg;
        unsigned lA = aBase + (unsigned)s * bufStride;
        unsigned lB = bBase + (unsigned)s * bufStride;
#if defined(__HIP_DEVICE_COMPILE__)
        asm volatile("global_load_async_to_lds_b128 %0, %1, off"
                     :: "v"(lA), "v"(gA) : "memory");
        asm volatile("global_load_async_to_lds_b128 %0, %1, off"
                     :: "v"(lB), "v"(gB) : "memory");
#else
        *(uint4*)&Ab[s][ro][seg] = *(const uint4*)gA;
        *(uint4*)&Bb[s][ro][seg] = *(const uint4*)gB;
#endif
    };

    v8f acc0 = {}, acc1 = {};
    stageAsync(0, 0);
    int cur = 0, nxt = 1;
    for (int c = 0; c < nchunks; ++c) {
        if (c + 1 < nchunks) stageAsync(c + 1, nxt);
#if defined(__HIP_DEVICE_COMPILE__)
        if (c + 1 < nchunks) asm volatile("s_wait_asynccnt 0x2" ::: "memory");
        else                 asm volatile("s_wait_asynccnt 0x0" ::: "memory");
#endif
        __syncthreads();    // all waves' chunk-c data landed in buffer `cur`
        v16bf a0, a1, bf;
        const __bf16* ar0 = &Ab[cur][mi * 32 + col][0];
        const __bf16* ar1 = &Ab[cur][mi * 32 + 16 + col][0];
        const __bf16* br  = &Bb[cur][ni * 16 + col][0];
        #pragma unroll
        for (int e = 0; e < 8; ++e) {       // A halves: lo lanes K 0..7,16..23
            a0[e]     = ar0[khalf * 8 + e];
            a0[e + 8] = ar0[16 + khalf * 8 + e];
            a1[e]     = ar1[khalf * 8 + e];
            a1[e + 8] = ar1[16 + khalf * 8 + e];
        }
        #pragma unroll
        for (int e = 0; e < 8; ++e) {       // B halves: lo lanes K 0..7,16..23
            bf[e]     = br[khalf * 8 + e];
            bf[e + 8] = br[16 + khalf * 8 + e];
        }
#if defined(__HIP_DEVICE_COMPILE__)
        acc0 = __builtin_amdgcn_wmma_f32_16x16x32_bf16(
            false, a0, false, bf, (short)0, acc0, false, false);
        acc1 = __builtin_amdgcn_wmma_f32_16x16x32_bf16(
            false, a1, false, bf, (short)0, acc1, false, false);
#else
        (void)a0; (void)a1; (void)bf;
#endif
        cur = nxt;
        nxt = (nxt == 2) ? 0 : nxt + 1;
    }

    int n = n0 + ni * 16 + col;
    if (n < Ntot) {
        int b = n / HW, px = n % HW;
        long obase = (long)b * Cout * HW + px;
        #pragma unroll
        for (int r = 0; r < 8; ++r) {
            int co  = co0 + mi * 32 + khalf * 8 + r;
            int co2 = co + 16;
            out[obase + (long)co  * HW] = acc0[r] + bias[co];
            out[obase + (long)co2 * HW] = acc1[r] + bias[co2];
        }
    }
}

// ---------------------------------------------------------------------------
// Bilinear x2 upsample (RGB skip path), half-pixel centers
// ---------------------------------------------------------------------------
__global__ __launch_bounds__(256) void upsample2_kernel(
    const float* __restrict__ in, float* __restrict__ out,
    int planes, int H, int W)
{
    int OH = H * 2, OW = W * 2;
    long total = (long)planes * OH * OW;
    long i = (long)blockIdx.x * 256 + threadIdx.x;
    if (i >= total) return;
    int ox = (int)(i % OW);
    long t = i / OW;
    int oy = (int)(t % OH);
    int p  = (int)(t / OH);
    float cx = ox * 0.5f - 0.25f;
    float cy = oy * 0.5f - 0.25f;
    int x0 = (int)floorf(cx), y0 = (int)floorf(cy);
    float fx = cx - x0, fy = cy - y0;
    int x1 = x0 + 1, y1 = y0 + 1;
    x0 = min(max(x0, 0), W - 1); x1 = min(max(x1, 0), W - 1);
    y0 = min(max(y0, 0), H - 1); y1 = min(max(y1, 0), H - 1);
    const float* ip = in + (long)p * H * W;
    float v = (1.f - fy) * ((1.f - fx) * ip[y0 * W + x0] + fx * ip[y0 * W + x1])
            +        fy  * ((1.f - fx) * ip[y1 * W + x0] + fx * ip[y1 * W + x1]);
    out[i] = v;
}

// ---------------------------------------------------------------------------
// toRGB: 1x1 modulated conv (Cout=3), optionally accumulating into img
// ---------------------------------------------------------------------------
__global__ __launch_bounds__(256) void torgb_kernel(
    const float* __restrict__ x, const float* __restrict__ w,
    const float* __restrict__ bias, const float* __restrict__ style,
    float* __restrict__ img, int Cin, int H, int W, int add)
{
    int n = blockIdx.x * 256 + threadIdx.x;
    int HW = H * W, Ntot = 8 * HW;
    if (n >= Ntot) return;
    int b = n / HW, px = n % HW;
    float a0 = bias[0], a1 = bias[1], a2 = bias[2];
    const float* xp = x + (long)b * Cin * HW + px;
    const float* sp = style + b * Cin;
    for (int ci = 0; ci < Cin; ++ci) {
        float v = xp[(long)ci * HW] * sp[ci];
        a0 += w[ci] * v;
        a1 += w[Cin + ci] * v;
        a2 += w[2 * Cin + ci] * v;
    }
    long base = (long)b * 3 * HW + px;
    if (add) {
        img[base]           += a0;
        img[base + HW]      += a1;
        img[base + 2L * HW] += a2;
    } else {
        img[base]           = a0;
        img[base + HW]      = a1;
        img[base + 2L * HW] = a2;
    }
}

// ---------------------------------------------------------------------------
// Host orchestration
// ---------------------------------------------------------------------------
extern "C" void kernel_launch(void* const* d_in, const int* in_sizes, int n_in,
                              void* d_out, int out_size, void* d_ws, size_t ws_size,
                              hipStream_t stream) {
    (void)in_sizes; (void)n_in; (void)out_size; (void)ws_size;

    const float* z = (const float*)d_in[0];
    MapParams mp;
    for (int i = 0; i < 8; ++i) {
        mp.w[i] = (const float*)d_in[1 + 2 * i];
        mp.b[i] = (const float*)d_in[2 + 2 * i];
    }
    const float* constx = (const float*)d_in[17];

    struct CP { const float *w, *b, *aw, *ab; };
    CP conv0[6], conv1[6], trgb[6];
    const int RESL[6] = {4, 8, 16, 32, 64, 128};
    const int CHL[6]  = {512, 512, 512, 512, 512, 256};
    int idx = 18;
    auto grab = [&](CP& c) {
        c.w  = (const float*)d_in[idx];
        c.b  = (const float*)d_in[idx + 1];
        c.aw = (const float*)d_in[idx + 2];
        c.ab = (const float*)d_in[idx + 3];
        idx += 4;
    };
    for (int i = 0; i < 6; ++i) {
        if (RESL[i] > 4) grab(conv0[i]);
        grab(conv1[i]);
        grab(trgb[i]);
    }

    // --- workspace carve ---
    char* wsp = (char*)d_ws;
    float*  wvec = (float*)wsp;                              // 16 KB
    float*  styl = (float*)(wsp + (16 << 10));               // 16 KB
    __bf16* wtb  = (__bf16*)(wsp + (32 << 10));              // 4.5 MB
    char* p = wsp + (32 << 10) + 9ull * 512 * 512 * sizeof(__bf16);
    p = (char*)(((uintptr_t)p + 255) & ~(uintptr_t)255);
    __bf16* xmod = (__bf16*)p;                               // 138.5 MB max
    p += 8ull * 130 * 130 * 512 * sizeof(__bf16);
    p = (char*)(((uintptr_t)p + 255) & ~(uintptr_t)255);
    const size_t BUF = 8ull * 256 * 128 * 128 * sizeof(float);  // 134 MB
    float* bufA = (float*)p; p += BUF;                       // x / conv1 out
    float* bufB = (float*)p; p += BUF;                       // conv0 out
    float* imgA = (float*)p; p += 8ull * 3 * 128 * 128 * sizeof(float);
    float* imgB = (float*)p;

    mapping_kernel<<<1, 1024, 0, stream>>>(z, mp, wvec);

    auto runStyle = [&](const CP& c, int cin) {
        int tot = 8 * cin;
        style_kernel<<<dim3((tot + 255) / 256), 256, 0, stream>>>(
            wvec, c.aw, c.ab, styl, cin);
    };
    auto runConv = [&](const float* in, long xbs, bool up, const CP& c,
                       int cin, int cout, int H, int W, float* outp) {
        runStyle(c, cin);
        long btot = 8L * cin * (2 * (W + 2) + 2 * H);
        fill_border_kernel<<<dim3((unsigned)((btot + 255) / 256)), 256, 0, stream>>>(
            xmod, cin, H, W);
        dim3 pg(8 * H, cin / 32);
        if (up)
            premod_up_kernel<<<pg, 256, 0, stream>>>(in, styl, xmod, cin, H, W);
        else
            premod_plain_kernel<<<pg, 256, 0, stream>>>(in, xbs, styl, xmod, cin, H, W);
        int wtot = cout * cin * 9;
        prep_weights_kernel<<<dim3((wtot + 255) / 256), 256, 0, stream>>>(
            c.w, wtb, cout, cin);
        int Ntot = 8 * H * W;
        dim3 g((Ntot + 63) / 64, cout / 64);
        modconv3x3_kernel<<<g, 256, 0, stream>>>(
            xmod, wtb, c.b, outp, cin, cout, H, W);
    };

    // res 4: x = const (batch-broadcast via xbs=0), conv1, torgb -> imgA
    runConv(constx, 0, false, conv1[0], 512, 512, 4, 4, bufA);
    runStyle(trgb[0], 512);
    torgb_kernel<<<dim3(1), 256, 0, stream>>>(
        bufA, trgb[0].w, trgb[0].b, styl, imgA, 512, 4, 4, 0);

    float* img = imgA;
    for (int i = 1; i < 6; ++i) {
        int res = RESL[i], cin = CHL[i - 1], cout = CHL[i];
        int H = res, W = res;
        // conv0: fused upsample of bufA (x at H/2) inside premod
        runConv(bufA, 0, true, conv0[i], cin, cout, H, W, bufB);
        // conv1: bufB -> bufA
        runConv(bufB, (long)cout * H * W, false, conv1[i], cout, cout, H, W, bufA);
        // skip-path RGB: upsample img, then add toRGB(bufA)
        float* imgd = (res == 128) ? (float*)d_out
                                   : ((img == imgA) ? imgB : imgA);
        long itot = 24L * H * W;
        upsample2_kernel<<<dim3((unsigned)((itot + 255) / 256)), 256, 0, stream>>>(
            img, imgd, 24, H / 2, W / 2);
        runStyle(trgb[i], cout);
        torgb_kernel<<<dim3((unsigned)((8L * H * W + 255) / 256)), 256, 0, stream>>>(
            bufA, trgb[i].w, trgb[i].b, styl, imgd, cout, H, W, 1);
        img = imgd;
    }
}
